// FluxTransformerBlock_88089779241101
// MI455X (gfx1250) — compile-verified, matrix operand order checked by reading, not run on hardware
//
#include <hip/hip_runtime.h>
#include <hip/hip_bf16.h>
#include <math.h>

// ---------------------------------------------------------------------------
// Flux dual-stream transformer block for MI455X (gfx1250, wave32, WMMA).
// All GEMMs run through a bf16 WMMA tile kernel (fp32 in/out, convert in LDS).
// Attention is a flash-style kernel with WMMA for QK^T and P*V.
// ---------------------------------------------------------------------------

typedef __bf16 bf16_t;
typedef __attribute__((ext_vector_type(16))) __bf16 v16bf;
typedef __attribute__((ext_vector_type(8)))  float  v8f;

#define D_MODEL 3072
#define S_IMG   2048
#define S_TXT   512
#define S_TOT   2560
#define NHEAD   24
#define DHEAD   128
#define FFDIM   12288
#define SIX_D   18432

// CDNA5 async global->LDS copy (ASYNCcnt). Probe-confirmed present; the
// builtin takes (v4i32 addrspace(1)*, v4i32 addrspace(3)*, imm offset, imm cpol).
#if defined(__gfx1250__) && \
    __has_builtin(__builtin_amdgcn_global_load_async_to_lds_b128) && \
    __has_builtin(__builtin_amdgcn_s_wait_asynccnt)
#define USE_ASYNC_LDS 1
typedef int async_v4i __attribute__((vector_size(16)));
typedef __attribute__((address_space(1))) async_v4i* g_v4i_ptr;
typedef __attribute__((address_space(3))) async_v4i* l_v4i_ptr;
#else
#define USE_ASYNC_LDS 0
#endif

__device__ inline v8f wmma_bf16(v16bf a, v16bf b, v8f c) {
  // D = A(16x32 bf16) * B(32x16 bf16) + C(16x16 f32)
  return __builtin_amdgcn_wmma_f32_16x16x32_bf16(false, a, false, b, (short)0, c,
                                                 false, false);
}

union FragU { v16bf v; uint4 u[2]; };

__device__ inline v16bf ld_frag(const bf16_t* p) {
  FragU r;
  r.u[0] = *reinterpret_cast<const uint4*>(p);
  r.u[1] = *reinterpret_cast<const uint4*>(p + 8);
  return r.v;
}

// ---------------------------------------------------------------------------
// Generic tiled GEMM: C[M,N] = act(A[M,K] * B[K,N] + bias)
// A,B fp32 in memory, converted to bf16 while staging into LDS.
// Block tile 128x128, K-step 32; 8 waves, each wave -> 32x64 (2x4 WMMA tiles).
// Software-pipelined: next K-tile's global loads issue before the WMMAs.
// head_split=1 scatters columns into a [H][S_TOT][DHEAD] layout for QKV.
// ---------------------------------------------------------------------------
__global__ __launch_bounds__(256) void flux_gemm(
    const float* __restrict__ A, const float* __restrict__ B,
    const float* __restrict__ bias, float* __restrict__ C,
    int M, int N, int K, int lda, int ldb, int ldc,
    int act_gelu, int head_split, int row_offset)
{
  __shared__ __align__(16) bf16_t As[128][40];   // [m][k], padded stride
  __shared__ __align__(16) bf16_t Bs[128][40];   // [n][k], padded stride

  const int tid  = threadIdx.x;
  const int wave = tid >> 5, lane = tid & 31;
  const int lh   = lane >> 4, ln = lane & 15;
  const int wm   = wave & 3,  wn = wave >> 2;
  const int m0   = blockIdx.y * 128, n0 = blockIdx.x * 128;

  const v8f vzero = {0.f,0.f,0.f,0.f,0.f,0.f,0.f,0.f};
  v8f acc[2][4];
  #pragma unroll
  for (int i = 0; i < 2; i++)
    #pragma unroll
    for (int j = 0; j < 4; j++) acc[i][j] = vzero;

  // A staging: thread owns (row, 16-wide k chunk); vector global loads.
  const int arow = tid >> 1, acol = (tid & 1) * 16;
  // B staging: thread owns one output column n and a 16-deep k chunk so the
  // transposed LDS write is 2x ds_store_b128 (column reads stay coalesced
  // across lanes).
  const int bn = tid & 127, bkg = (tid >> 7) * 16;

  float4 areg[4];
  float  breg[16];

  // prologue: fetch K-tile 0 into registers
  {
    const float* ap = A + (size_t)(m0 + arow) * lda + acol;
    #pragma unroll
    for (int v = 0; v < 4; v++) areg[v] = reinterpret_cast<const float4*>(ap)[v];
    const float* bp = B + (size_t)bkg * ldb + n0 + bn;
    #pragma unroll
    for (int k = 0; k < 16; k++) breg[k] = bp[(size_t)k * ldb];
  }

  for (int k0 = 0; k0 < K; k0 += 32) {
    // commit registers -> LDS (convert fp32 -> bf16)
    #pragma unroll
    for (int v = 0; v < 4; v++) {
      As[arow][acol + v*4 + 0] = (bf16_t)areg[v].x;
      As[arow][acol + v*4 + 1] = (bf16_t)areg[v].y;
      As[arow][acol + v*4 + 2] = (bf16_t)areg[v].z;
      As[arow][acol + v*4 + 3] = (bf16_t)areg[v].w;
    }
    {
      __align__(16) bf16_t tmp[16];
      #pragma unroll
      for (int k = 0; k < 16; k++) tmp[k] = (bf16_t)breg[k];
      *reinterpret_cast<uint4*>(&Bs[bn][bkg + 0]) = *reinterpret_cast<const uint4*>(&tmp[0]);
      *reinterpret_cast<uint4*>(&Bs[bn][bkg + 8]) = *reinterpret_cast<const uint4*>(&tmp[8]);
    }
    __syncthreads();

    // software pipeline: issue next tile's global loads before the WMMAs
    if (k0 + 32 < K) {
      const float* ap = A + (size_t)(m0 + arow) * lda + (k0 + 32) + acol;
      #pragma unroll
      for (int v = 0; v < 4; v++) areg[v] = reinterpret_cast<const float4*>(ap)[v];
      const float* bp = B + (size_t)(k0 + 32 + bkg) * ldb + n0 + bn;
      #pragma unroll
      for (int k = 0; k < 16; k++) breg[k] = bp[(size_t)k * ldb];
      if (k0 + 64 < K) {
        __builtin_prefetch(ap + 32, 0, 1);
        __builtin_prefetch(bp + (size_t)32 * ldb, 0, 1);
      }
    }

    v16bf af0 = ld_frag(&As[wm*32 +  0 + ln][lh*16]);
    v16bf af1 = ld_frag(&As[wm*32 + 16 + ln][lh*16]);
    #pragma unroll
    for (int nt = 0; nt < 4; nt++) {
      v16bf bf = ld_frag(&Bs[wn*64 + nt*16 + ln][lh*16]);
      acc[0][nt] = wmma_bf16(af0, bf, acc[0][nt]);
      acc[1][nt] = wmma_bf16(af1, bf, acc[1][nt]);
    }
    __syncthreads();
  }

  #pragma unroll
  for (int mt = 0; mt < 2; mt++)
    #pragma unroll
    for (int nt = 0; nt < 4; nt++)
      #pragma unroll
      for (int e = 0; e < 8; e++) {
        int r = m0 + wm*32 + mt*16 + e + lh*8;
        int c = n0 + wn*64 + nt*16 + ln;
        float v = acc[mt][nt][e];
        if (bias) v += bias[c];
        if (act_gelu) {
          float x = v;
          v = 0.5f * x * (1.0f + tanhf(0.7978845608f * (x + 0.044715f*x*x*x)));
        }
        if (!head_split) {
          C[(size_t)r * ldc + c] = v;
        } else {
          int h = c >> 7, d = c & 127;
          C[((size_t)h * S_TOT + row_offset + r) * DHEAD + d] = v;
        }
      }
}

// ---------------------------------------------------------------------------
// adaLN: st = silu(temb); mod = st @ W + b for both adaln and adaln_c.
// ---------------------------------------------------------------------------
__global__ __launch_bounds__(256) void flux_adaln(
    const float* __restrict__ temb,
    const float* __restrict__ W0, const float* __restrict__ b0,
    const float* __restrict__ W1, const float* __restrict__ b1,
    float* __restrict__ mod, float* __restrict__ modc)
{
  int j = blockIdx.x * 256 + threadIdx.x;            // 0..2*SIX_D-1
  int col; const float *W, *b; float* out;
  if (j < SIX_D) { col = j;          W = W0; b = b0; out = mod;  }
  else           { col = j - SIX_D;  W = W1; b = b1; out = modc; }
  float acc = 0.f;
  for (int i = 0; i < D_MODEL; i++) {
    float t = temb[i];
    float s = t / (1.f + __expf(-t));
    acc += s * W[(size_t)i * SIX_D + col];
  }
  out[col] = acc + b[col];
}

// ---------------------------------------------------------------------------
// LayerNorm + adaLN modulation: Y = LN(X)*(1+scale) + shift   (one row/block)
// ---------------------------------------------------------------------------
__global__ __launch_bounds__(256) void flux_ln_mod(
    const float* __restrict__ X, float* __restrict__ Y,
    const float* __restrict__ scale, const float* __restrict__ shift)
{
  __shared__ float r1[256], r2[256];
  const int row = blockIdx.x, tid = threadIdx.x;
  const float* xr = X + (size_t)row * D_MODEL;
  float xl[12], s = 0.f, s2 = 0.f;
  #pragma unroll
  for (int i = 0; i < 12; i++) {
    float x = xr[tid + i*256]; xl[i] = x; s += x; s2 += x*x;
  }
  r1[tid] = s; r2[tid] = s2; __syncthreads();
  for (int st = 128; st > 0; st >>= 1) {
    if (tid < st) { r1[tid] += r1[tid+st]; r2[tid] += r2[tid+st]; }
    __syncthreads();
  }
  float mu  = r1[0] * (1.f / D_MODEL);
  float var = r2[0] * (1.f / D_MODEL) - mu*mu;
  float inv = rsqrtf(var + 1e-6f);
  float* yr = Y + (size_t)row * D_MODEL;
  #pragma unroll
  for (int i = 0; i < 12; i++) {
    int c = tid + i*256;
    yr[c] = (xl[i] - mu) * inv * (1.f + scale[c]) + shift[c];
  }
}

// ---------------------------------------------------------------------------
// Per-head RMSNorm + RoPE on q,k; outputs bf16 for the attention WMMAs.
// One block per (head, token); 128 threads (one per d).
// ---------------------------------------------------------------------------
__global__ __launch_bounds__(128) void flux_rmsrope(
    const float* __restrict__ QR, const float* __restrict__ KR,
    bf16_t* __restrict__ QB, bf16_t* __restrict__ KB,
    const float* __restrict__ rq,  const float* __restrict__ rk,
    const float* __restrict__ rqa, const float* __restrict__ rka,
    const float* __restrict__ cosT, const float* __restrict__ sinT)
{
  __shared__ float sq[128], sk[128], red[128];
  const int idx = blockIdx.x;
  const int h = idx / S_TOT, s = idx - h * S_TOT;
  const int d = threadIdx.x;
  const size_t base = ((size_t)h * S_TOT + s) * DHEAD;
  float qv = QR[base + d], kv = KR[base + d];

  red[d] = qv * qv; __syncthreads();
  for (int st = 64; st > 0; st >>= 1) { if (d < st) red[d] += red[d+st]; __syncthreads(); }
  float qinv = rsqrtf(red[0] * (1.f/DHEAD) + 1e-6f);
  __syncthreads();
  red[d] = kv * kv; __syncthreads();
  for (int st = 64; st > 0; st >>= 1) { if (d < st) red[d] += red[d+st]; __syncthreads(); }
  float kinv = rsqrtf(red[0] * (1.f/DHEAD) + 1e-6f);
  __syncthreads();

  const float* wq = (s < S_TXT) ? rqa : rq;
  const float* wk = (s < S_TXT) ? rka : rk;
  float qn = qv * qinv * wq[d];
  float kn = kv * kinv * wk[d];
  sq[d] = qn; sk[d] = kn; __syncthreads();

  float c  = cosT[(size_t)s * DHEAD + d];
  float sn = sinT[(size_t)s * DHEAD + d];
  float qo, ko;
  if ((d & 1) == 0) { qo = qn*c - sq[d+1]*sn; ko = kn*c - sk[d+1]*sn; }
  else              { qo = qn*c + sq[d-1]*sn; ko = kn*c + sk[d-1]*sn; }
  QB[base + d] = (bf16_t)qo;
  KB[base + d] = (bf16_t)ko;
}

// ---------------------------------------------------------------------------
// Flash attention: Br=128 queries/block (8 waves x 16 rows), Bc=64 keys/iter.
// Q,K bf16 [H][S][DH]; V fp32 [H][S][DH] (converted while staging).
// K tile staged with CDNA5 async global->LDS copies (ASYNCcnt).
// Output fp32 [S][H*DH] row-major so the O-projection GEMM consumes it.
// ---------------------------------------------------------------------------
__global__ __launch_bounds__(256) void flux_flash(
    const bf16_t* __restrict__ Q, const bf16_t* __restrict__ Kb,
    const float* __restrict__ V, float* __restrict__ O)
{
  __shared__ __align__(16) bf16_t Ks[64][136];     // [key][d], padded
  __shared__ __align__(16) bf16_t Vs[128][80];     // [d][key], padded
  __shared__ __align__(16) bf16_t Ps[8][16][80];   // per-wave P tile

  const int h = blockIdx.x, mblk = blockIdx.y;
  const int tid = threadIdx.x;
  const int wave = tid >> 5, lane = tid & 31;
  const int lh = lane >> 4, ln = lane & 15;
  const float scale = 0.0883883476f;               // 1/sqrt(128)

  const int qrow = mblk * 128 + wave * 16 + ln;
  const bf16_t* qptr = Q + ((size_t)h * S_TOT + qrow) * DHEAD;
  v16bf qf[4];
  #pragma unroll
  for (int c = 0; c < 4; c++) qf[c] = ld_frag(qptr + c*32 + lh*16);

  const v8f vzero = {0.f,0.f,0.f,0.f,0.f,0.f,0.f,0.f};
  v8f o[8];
  #pragma unroll
  for (int t = 0; t < 8; t++) o[t] = vzero;
  float m_i[8], l_i[8];
  #pragma unroll
  for (int e = 0; e < 8; e++) { m_i[e] = -1e30f; l_i[e] = 0.f; }

  const int vd = tid & 127, vkg = (tid >> 7) * 32;   // V staging ownership

  for (int kb = 0; kb < S_TOT; kb += 64) {
    // ---- stage K tile (bf16 -> bf16 copy, 16B chunks per lane) ----
#if USE_ASYNC_LDS
    for (int i = tid; i < 1024; i += 256) {
      int row = i >> 4, off = (i & 15) * 8;
      const bf16_t* src = Kb + ((size_t)h * S_TOT + kb + row) * DHEAD + off;
      __builtin_amdgcn_global_load_async_to_lds_b128(
          (g_v4i_ptr)(uintptr_t)src,
          (l_v4i_ptr)(unsigned)(uintptr_t)&Ks[row][off], 0, 0);
    }
#else
    for (int i = tid; i < 1024; i += 256) {
      int row = i >> 4, off = (i & 15) * 8;
      *reinterpret_cast<uint4*>(&Ks[row][off]) =
        *reinterpret_cast<const uint4*>(Kb + ((size_t)h * S_TOT + kb + row) * DHEAD + off);
    }
#endif
    // ---- stage V tile transposed: thread owns column d, 32 keys ----
    {
      __align__(16) bf16_t vt[32];
      #pragma unroll
      for (int k = 0; k < 32; k++)
        vt[k] = (bf16_t)V[((size_t)h * S_TOT + kb + vkg + k) * DHEAD + vd];
      #pragma unroll
      for (int c = 0; c < 4; c++)
        *reinterpret_cast<uint4*>(&Vs[vd][vkg + c*8]) =
            *reinterpret_cast<const uint4*>(&vt[c*8]);
    }
#if USE_ASYNC_LDS
    __builtin_amdgcn_s_wait_asynccnt(0);
#endif
    __syncthreads();

    // ---- scores S = q . k^T  (16 x 64 per wave) ----
    v8f sacc[4];
    #pragma unroll
    for (int nt = 0; nt < 4; nt++) sacc[nt] = vzero;
    #pragma unroll
    for (int nt = 0; nt < 4; nt++)
      #pragma unroll
      for (int c = 0; c < 4; c++) {
        v16bf kf = ld_frag(&Ks[nt*16 + ln][c*32 + lh*16]);
        sacc[nt] = wmma_bf16(qf[c], kf, sacc[nt]);
      }

    // ---- online softmax (rows e / e+8 per lane-half; 16-lane reductions) ----
    float alpha[8];
    #pragma unroll
    for (int e = 0; e < 8; e++) {
      float mx = sacc[0][e];
      mx = fmaxf(mx, sacc[1][e]); mx = fmaxf(mx, sacc[2][e]); mx = fmaxf(mx, sacc[3][e]);
      for (int off = 1; off < 16; off <<= 1) mx = fmaxf(mx, __shfl_xor(mx, off, 32));
      float mnew = fmaxf(m_i[e], mx * scale);
      alpha[e] = __expf(m_i[e] - mnew);
      m_i[e] = mnew;
    }
    #pragma unroll
    for (int nt = 0; nt < 4; nt++)
      #pragma unroll
      for (int e = 0; e < 8; e++)
        sacc[nt][e] = __expf(sacc[nt][e] * scale - m_i[e]);
    #pragma unroll
    for (int e = 0; e < 8; e++) {
      float rs = sacc[0][e] + sacc[1][e] + sacc[2][e] + sacc[3][e];
      for (int off = 1; off < 16; off <<= 1) rs += __shfl_xor(rs, off, 32);
      l_i[e] = l_i[e] * alpha[e] + rs;
    }
    #pragma unroll
    for (int t = 0; t < 8; t++)
      #pragma unroll
      for (int e = 0; e < 8; e++) o[t][e] *= alpha[e];
    #pragma unroll
    for (int nt = 0; nt < 4; nt++)
      #pragma unroll
      for (int e = 0; e < 8; e++)
        Ps[wave][e + 8*lh][nt*16 + ln] = (bf16_t)sacc[nt][e];
    __syncthreads();

    // ---- O += P * V   (A = P 16x64, B = V 64x16 per d-tile) ----
    v16bf pf0 = ld_frag(&Ps[wave][ln][ 0 + lh*16]);
    v16bf pf1 = ld_frag(&Ps[wave][ln][32 + lh*16]);
    #pragma unroll
    for (int dt = 0; dt < 8; dt++) {
      v16bf vf0 = ld_frag(&Vs[dt*16 + ln][ 0 + lh*16]);
      v16bf vf1 = ld_frag(&Vs[dt*16 + ln][32 + lh*16]);
      o[dt] = wmma_bf16(pf0, vf0, o[dt]);
      o[dt] = wmma_bf16(pf1, vf1, o[dt]);
    }
    __syncthreads();
  }

  #pragma unroll
  for (int dt = 0; dt < 8; dt++)
    #pragma unroll
    for (int e = 0; e < 8; e++) {
      int row = mblk*128 + wave*16 + e + 8*lh;
      int col = h*DHEAD + dt*16 + ln;
      O[(size_t)row * D_MODEL + col] = o[dt][e] / l_i[e];
    }
}

// --------------------------- elementwise helpers ---------------------------
__global__ __launch_bounds__(256) void flux_resgate(
    const float* __restrict__ X, const float* __restrict__ Yv,
    const float* __restrict__ gate, float* __restrict__ out, int n)
{
  int i = blockIdx.x * 256 + threadIdx.x;
  if (i < n) { int c = i % D_MODEL; out[i] = X[i] + gate[c] * Yv[i]; }
}

__global__ __launch_bounds__(256) void flux_meancond(
    const float* __restrict__ E, const float* __restrict__ temb,
    float* __restrict__ cond)
{
  int c = blockIdx.x * 256 + threadIdx.x;
  float s = 0.f;
  for (int r = 0; r < S_TXT; r++) s += E[(size_t)r * D_MODEL + c];
  cond[c] = temb[c] + s * (1.f / S_TXT);
}

__global__ __launch_bounds__(256) void flux_addvec(
    const float* __restrict__ X, const float* __restrict__ v,
    float* __restrict__ out, int n)
{
  int i = blockIdx.x * 256 + threadIdx.x;
  if (i < n) out[i] = X[i] + v[i % D_MODEL];
}

__global__ __launch_bounds__(256) void flux_mask(
    const float* __restrict__ tsact, const float* __restrict__ w2,
    const float* __restrict__ b2, float* __restrict__ mask)
{
  int tok = blockIdx.x * 8 + (threadIdx.x >> 5);
  int lane = threadIdx.x & 31;
  float acc = 0.f;
  for (int j = lane; j < 128; j += 32) {
    float x = tsact[(size_t)tok * 128 + j];
    x = x / (1.f + __expf(-x));
    acc += x * w2[j];
  }
  for (int off = 16; off > 0; off >>= 1) acc += __shfl_xor(acc, off, 32);
  if (lane == 0) mask[tok] = ((acc + b2[0]) > 0.f) ? 1.f : 0.f;
}

__global__ __launch_bounds__(256) void flux_final(
    const float* __restrict__ base, const float* __restrict__ ff,
    const float* __restrict__ gate, const float* __restrict__ mask,
    float* __restrict__ out, int n)
{
  int i = blockIdx.x * 256 + threadIdx.x;
  if (i >= n) return;
  int r = i / D_MODEL, c = i - r * D_MODEL;
  float mv = mask ? mask[r] : 1.f;
  out[i] = base[i] + gate[c] * (mv * ff[i]);
}

// ---------------------------------------------------------------------------
// Orchestration
// ---------------------------------------------------------------------------
static inline void launch_gemm(hipStream_t st, const float* A, const float* B,
                               const float* bias, float* C, int M, int N, int K,
                               int lda, int ldb, int ldc, int gelu,
                               int head_split, int row_offset) {
  dim3 g(N / 128, M / 128), b(256);
  flux_gemm<<<g, b, 0, st>>>(A, B, bias, C, M, N, K, lda, ldb, ldc,
                             gelu, head_split, row_offset);
}

extern "C" void kernel_launch(void* const* d_in, const int* in_sizes, int n_in,
                              void* d_out, int out_size, void* d_ws, size_t ws_size,
                              hipStream_t stream) {
  const float* hid   = (const float*)d_in[0];   // [2048][3072]
  const float* enc   = (const float*)d_in[1];   // [512][3072]
  const float* temb  = (const float*)d_in[2];   // [3072]
  const float* cosT  = (const float*)d_in[3];   // [2560][128]
  const float* sinT  = (const float*)d_in[4];   // [2560][128]
  const float* adaln_w   = (const float*)d_in[5];
  const float* adaln_b   = (const float*)d_in[6];
  const float* adaln_c_w = (const float*)d_in[7];
  const float* adaln_c_b = (const float*)d_in[8];
  const float* Wq  = (const float*)d_in[9];   const float* bq  = (const float*)d_in[10];
  const float* Wqa = (const float*)d_in[11];  const float* bqa = (const float*)d_in[12];
  const float* Wk  = (const float*)d_in[13];  const float* bk  = (const float*)d_in[14];
  const float* Wka = (const float*)d_in[15];  const float* bka = (const float*)d_in[16];
  const float* Wv  = (const float*)d_in[17];  const float* bv  = (const float*)d_in[18];
  const float* Wva = (const float*)d_in[19];  const float* bva = (const float*)d_in[20];
  const float* rms_q  = (const float*)d_in[21];
  const float* rms_k  = (const float*)d_in[22];
  const float* rms_qa = (const float*)d_in[23];
  const float* rms_ka = (const float*)d_in[24];
  const float* Wo  = (const float*)d_in[25];  const float* bo  = (const float*)d_in[26];
  const float* Woa = (const float*)d_in[27];  const float* boa = (const float*)d_in[28];
  const float* ff_w1  = (const float*)d_in[29]; const float* ff_b1  = (const float*)d_in[30];
  const float* ff_w2  = (const float*)d_in[31]; const float* ff_b2  = (const float*)d_in[32];
  const float* ffc_w1 = (const float*)d_in[33]; const float* ffc_b1 = (const float*)d_in[34];
  const float* ffc_w2 = (const float*)d_in[35]; const float* ffc_b2 = (const float*)d_in[36];
  const float* ts_w1 = (const float*)d_in[37]; const float* ts_b1 = (const float*)d_in[38];
  const float* ts_w2 = (const float*)d_in[39]; const float* ts_b2 = (const float*)d_in[40];

  float* ws = (float*)d_ws;
  // workspace layout (float units), with deliberate reuse of dead buffers
  const size_t OFF_MOD   = 0;
  const size_t OFF_MODC  = 18432;
  const size_t OFF_COND  = 36864;
  const size_t OFF_MASK  = 39936;
  const size_t OFF_NX    = 65536;                       // 2048*3072 (also feat / nh)
  const size_t OFF_NC    = OFF_NX   + 6291456;          // 512*3072  (also ne)
  const size_t OFF_QRAW  = OFF_NC   + 1572864;          // 24*2560*128
  const size_t OFF_KRAW  = OFF_QRAW + 7864320;
  const size_t OFF_VRAW  = OFF_KRAW + 7864320;
  const size_t OFF_QBF   = OFF_VRAW + 7864320;          // bf16: 7864320 elems
  const size_t OFF_KBF   = OFF_QBF  + 3932160;
  const size_t OFF_ATTN  = OFF_KBF  + 3932160;          // 2560*3072
  const size_t OFF_OPROJ = OFF_ATTN + 7864320;          // 2560*3072
  // overlays (after attention / O-proj are consumed):
  const size_t OFF_FF1   = OFF_QRAW;                    // 2048*12288
  const size_t OFF_FF2   = OFF_FF1  + 25165824;         // 2048*3072
  const size_t OFF_FFC1  = OFF_FF2  + 6291456;          // 512*12288
  const size_t OFF_FFC2  = OFF_FFC1 + 6291456;          // 512*3072
  const size_t OFF_TSACT = OFF_OPROJ;                   // 2048*128 (after resgate)
  const size_t OFF_HBUF  = OFF_OPROJ + 7864320;         // 2048*3072
  const size_t OFF_EBUF  = OFF_HBUF + 6291456;          // 512*3072

  float* mod   = ws + OFF_MOD;
  float* modc  = ws + OFF_MODC;
  float* cond  = ws + OFF_COND;
  float* maskv = ws + OFF_MASK;
  float* nx    = ws + OFF_NX;
  float* nc    = ws + OFF_NC;
  float* qraw  = ws + OFF_QRAW;
  float* kraw  = ws + OFF_KRAW;
  float* vraw  = ws + OFF_VRAW;
  bf16_t* qbf  = (bf16_t*)(ws + OFF_QBF);
  bf16_t* kbf  = (bf16_t*)(ws + OFF_KBF);
  float* attn  = ws + OFF_ATTN;
  float* oproj = ws + OFF_OPROJ;
  float* ff1   = ws + OFF_FF1;
  float* ff2   = ws + OFF_FF2;
  float* ffc1  = ws + OFF_FFC1;
  float* ffc2  = ws + OFF_FFC2;
  float* tsact = ws + OFF_TSACT;
  float* hbuf  = ws + OFF_HBUF;
  float* ebuf  = ws + OFF_EBUF;

  float* out_enc = (float*)d_out;                       // [512][3072]
  float* out_hid = (float*)d_out + (size_t)S_TXT * D_MODEL;

  // 1) adaLN modulation vectors
  flux_adaln<<<dim3(2 * SIX_D / 256), dim3(256), 0, stream>>>(
      temb, adaln_w, adaln_b, adaln_c_w, adaln_c_b, mod, modc);

  // 2) LN + msa modulation
  flux_ln_mod<<<dim3(S_IMG), dim3(256), 0, stream>>>(hid, nx, mod  + 1*D_MODEL, mod  + 0*D_MODEL);
  flux_ln_mod<<<dim3(S_TXT), dim3(256), 0, stream>>>(enc, nc, modc + 1*D_MODEL, modc + 0*D_MODEL);

  // 3) QKV projections (ctx rows 0..511, img rows 512..2559, head-split layout)
  launch_gemm(stream, nc, Wqa, bqa, qraw, S_TXT, D_MODEL, D_MODEL, D_MODEL, D_MODEL, 0, 0, 1, 0);
  launch_gemm(stream, nc, Wka, bka, kraw, S_TXT, D_MODEL, D_MODEL, D_MODEL, D_MODEL, 0, 0, 1, 0);
  launch_gemm(stream, nc, Wva, bva, vraw, S_TXT, D_MODEL, D_MODEL, D_MODEL, D_MODEL, 0, 0, 1, 0);
  launch_gemm(stream, nx, Wq,  bq,  qraw, S_IMG, D_MODEL, D_MODEL, D_MODEL, D_MODEL, 0, 0, 1, S_TXT);
  launch_gemm(stream, nx, Wk,  bk,  kraw, S_IMG, D_MODEL, D_MODEL, D_MODEL, D_MODEL, 0, 0, 1, S_TXT);
  launch_gemm(stream, nx, Wv,  bv,  vraw, S_IMG, D_MODEL, D_MODEL, D_MODEL, D_MODEL, 0, 0, 1, S_TXT);

  // 4) RMSNorm + RoPE -> bf16 q,k
  flux_rmsrope<<<dim3(NHEAD * S_TOT), dim3(128), 0, stream>>>(
      qraw, kraw, qbf, kbf, rms_q, rms_k, rms_qa, rms_ka, cosT, sinT);

  // 5) flash attention
  flux_flash<<<dim3(NHEAD, S_TOT / 128), dim3(256), 0, stream>>>(qbf, kbf, vraw, attn);

  // 6) output projections
  launch_gemm(stream, attn, Woa, boa, oproj, S_TXT, D_MODEL, D_MODEL,
              D_MODEL, D_MODEL, D_MODEL, 0, 0, 0);
  launch_gemm(stream, attn + (size_t)S_TXT * D_MODEL, Wo, bo,
              oproj + (size_t)S_TXT * D_MODEL, S_IMG, D_MODEL, D_MODEL,
              D_MODEL, D_MODEL, D_MODEL, 0, 0, 0);

  // 7) gated residuals
  {
    int n = S_IMG * D_MODEL;
    flux_resgate<<<dim3((n + 255)/256), dim3(256), 0, stream>>>(
        hid, oproj + (size_t)S_TXT * D_MODEL, mod + 2*D_MODEL, hbuf, n);
    n = S_TXT * D_MODEL;
    flux_resgate<<<dim3((n + 255)/256), dim3(256), 0, stream>>>(
        enc, oproj, modc + 2*D_MODEL, ebuf, n);
  }

  // 8) cond = temb + mean(ebuf); feat = hbuf + cond
  flux_meancond<<<dim3(D_MODEL / 256), dim3(256), 0, stream>>>(ebuf, temb, cond);
  {
    int n = S_IMG * D_MODEL;
    flux_addvec<<<dim3((n + 255)/256), dim3(256), 0, stream>>>(hbuf, cond, nx, n);
  }

  // 9) token-select: tsact = feat @ ts_w1 + ts_b1 ; mask from silu+dot
  launch_gemm(stream, nx, ts_w1, ts_b1, tsact, S_IMG, 128, D_MODEL,
              D_MODEL, 128, 128, 0, 0, 0);
  flux_mask<<<dim3(S_IMG / 8), dim3(256), 0, stream>>>(tsact, ts_w2, ts_b2, maskv);

  // 10) LN + mlp modulation
  flux_ln_mod<<<dim3(S_IMG), dim3(256), 0, stream>>>(hbuf, nx, mod  + 4*D_MODEL, mod  + 3*D_MODEL);
  flux_ln_mod<<<dim3(S_TXT), dim3(256), 0, stream>>>(ebuf, nc, modc + 4*D_MODEL, modc + 3*D_MODEL);

  // 11) feed-forward (gelu on first GEMM)
  launch_gemm(stream, nx,   ff_w1,  ff_b1,  ff1,  S_IMG, FFDIM,  D_MODEL, D_MODEL, FFDIM,  FFDIM,  1, 0, 0);
  launch_gemm(stream, ff1,  ff_w2,  ff_b2,  ff2,  S_IMG, D_MODEL, FFDIM,  FFDIM,  D_MODEL, D_MODEL, 0, 0, 0);
  launch_gemm(stream, nc,   ffc_w1, ffc_b1, ffc1, S_TXT, FFDIM,  D_MODEL, D_MODEL, FFDIM,  FFDIM,  1, 0, 0);
  launch_gemm(stream, ffc1, ffc_w2, ffc_b2, ffc2, S_TXT, D_MODEL, FFDIM,  FFDIM,  D_MODEL, D_MODEL, 0, 0, 0);

  // 12) final gated residuals -> d_out (encoder first, then hidden)
  {
    int n = S_TXT * D_MODEL;
    flux_final<<<dim3((n + 255)/256), dim3(256), 0, stream>>>(
        ebuf, ffc2, modc + 5*D_MODEL, nullptr, out_enc, n);
    n = S_IMG * D_MODEL;
    flux_final<<<dim3((n + 255)/256), dim3(256), 0, stream>>>(
        hbuf, ff2, mod + 5*D_MODEL, maskv, out_hid, n);
  }
  (void)in_sizes; (void)n_in; (void)out_size; (void)ws_size;
}
